// SnG_58591943852541
// MI455X (gfx1250) — matrix operand order repaired and checked by reading, not compiled, not run on hardware
//
#include <hip/hip_runtime.h>

// ---------------------------------------------------------------------------
// CDNA5 / gfx1250 GraphDTA implementation, v3.
// Dense matmuls -> v_wmma_f32_16x16x32_f16, LDS-free:
//   A fragment = 4x global_load_b128 (f32) + v_cvt_pk (rows/K-runs map 1:1 to lanes)
//   B fragment = pre-packed f16 in fragment order -> 2x global_load_b128
// Edge scatter: one wave32 per edge, feature = lane (coalesced load + atomics).
// All tile dims exact multiples (M%64==0, N%32==0, K padded to 32) -> no guards.
// ---------------------------------------------------------------------------

typedef __attribute__((ext_vector_type(16))) _Float16 v16h;
typedef __attribute__((ext_vector_type(8)))  float    v8f;

#define CDIV(a, b) (((a) + (b) - 1) / (b))

// ---- weight pre-pack: f32 [K x N] -> f16 fragment-ordered [Kp/32][N/16][32][16]
// element (tile kt,nt, lane l, slot j) = W[kt*32 + (l>=16?8:0) + (j<8?j:j+8), nt*16 + (l&15)]
// convmode: W is conv_w [32,100,8]; logical W[kidx][o] = conv_w[o, kidx>>3, kidx&7]
__global__ void k_prepack(const float* __restrict__ W, _Float16* __restrict__ out,
                          int K, int N, int total, int convmode)
{
    int idx = blockIdx.x * blockDim.x + threadIdx.x;
    if (idx >= total) return;
    int j    = idx & 15;
    int l    = (idx >> 4) & 31;
    int tile = idx >> 9;
    int NT   = N >> 4;
    int nt   = tile % NT, kt = tile / NT;
    int n    = (nt << 4) + (l & 15);
    int kk   = (kt << 5) + ((l & 16) ? 8 : 0) + (j < 8 ? j : j + 8);
    float v = 0.f;
    if (kk < K)
        v = convmode ? W[(n * 100 + (kk >> 3)) * 8 + (kk & 7)]
                     : W[(size_t)kk * N + n];
    out[idx] = (_Float16)v;
}

// ---- generic GEMM: C[M,N] = act(A[M,K] @ W + bias), A f32 (lda >= Kp, pad=0),
// W pre-packed f16. Wave = one 16x16 tile; block = 8 waves = 64x32 block tile.
__global__ void __launch_bounds__(256)
k_gemm(const float* __restrict__ A, int lda, const _Float16* __restrict__ Wp,
       const float* __restrict__ bias, float* __restrict__ C, int ldc,
       int N, int K, int relu)
{
    const int w    = threadIdx.x >> 5;
    const int lane = threadIdx.x & 31;
    const int M0 = blockIdx.x * 64 + (w >> 1) * 16;
    const int N0 = blockIdx.y * 32 + (w & 1) * 16;
    const int NT = N >> 4, nt = N0 >> 4;
    const int r    = lane & 15;
    const int koA  = (lane & 16) ? 8 : 0;       // K-run base: {0,16..} or {8,24..}
    const float* arow = A + (size_t)(M0 + r) * lda + koA;

    v8f c = {};
    const int KT = (K + 31) >> 5;
    for (int kt = 0; kt < KT; ++kt) {
        const float4* pa = (const float4*)(arow + (kt << 5));   // 16B aligned
        float4 a0 = pa[0], a1 = pa[1];          // K = koA+0..7
        float4 a2 = pa[4], a3 = pa[5];          // K = koA+16..23
        v16h a;
        a[0] = (_Float16)a0.x; a[1] = (_Float16)a0.y;
        a[2] = (_Float16)a0.z; a[3] = (_Float16)a0.w;
        a[4] = (_Float16)a1.x; a[5] = (_Float16)a1.y;
        a[6] = (_Float16)a1.z; a[7] = (_Float16)a1.w;
        a[8]  = (_Float16)a2.x; a[9]  = (_Float16)a2.y;
        a[10] = (_Float16)a2.z; a[11] = (_Float16)a2.w;
        a[12] = (_Float16)a3.x; a[13] = (_Float16)a3.y;
        a[14] = (_Float16)a3.z; a[15] = (_Float16)a3.w;
        const v16h b = *(const v16h*)(Wp + (((size_t)(kt * NT + nt) * 32 + lane) << 4));
        c = __builtin_amdgcn_wmma_f32_16x16x32_f16(false, a, false, b, (short)0, c,
                                                   false, false);
    }
    const int gn = N0 + (lane & 15);
    const int mb = (lane & 16) ? 8 : 0;
    const float bv = bias[gn];
#pragma unroll
    for (int rr = 0; rr < 8; ++rr) {
        float v = c[rr] + bv;
        if (relu) v = v > 0.f ? v : 0.f;
        C[(size_t)(M0 + mb + rr) * ldc + gn] = v;
    }
}

// ---- drug branch: embedding gather to f16 ---------------------------------
__global__ void k_embed(const int* __restrict__ xd, const float* __restrict__ emb,
                        _Float16* __restrict__ E16, int total)
{
    int idx = blockIdx.x * blockDim.x + threadIdx.x;   // = (b*100+i)*128 + t
    if (idx >= total) return;
    int t   = idx & 127;
    int tok = xd[idx >> 7];
    E16[idx] = (_Float16)emb[tok * 128 + t];
}

// ---- conv1d as implicit GEMM: rows (b,t), K = 800 = 25*32, N = 32 ---------
// A element (row=(b,t), kidx=(i,k)) = E16[b, i, t+k]; t+k <= 127 always valid.
__global__ void __launch_bounds__(256)
k_conv_gemm(const _Float16* __restrict__ E16, const _Float16* __restrict__ Wp,
            const float* __restrict__ cb, float* __restrict__ out)
{
    const int w    = threadIdx.x >> 5;
    const int lane = threadIdx.x & 31;
    const int M0 = blockIdx.x * 64 + (w >> 1) * 16;
    const int nt = w & 1;
    const int r    = lane & 15;
    const int koA  = (lane & 16) ? 8 : 0;
    const int row  = M0 + r;
    const int bb = row / 121, t = row % 121;
    const _Float16* eb = E16 + (((size_t)bb * 100) << 7) + t;

    v8f c = {};
    for (int kt = 0; kt < 25; ++kt) {
        const int k1 = (kt << 5) + koA;                    // multiple of 8
        const _Float16* p1 = eb + ((size_t)(k1 >> 3) << 7);        // K = k1..k1+7
        const _Float16* p2 = eb + ((size_t)((k1 + 16) >> 3) << 7); // K = k1+16..k1+23
        v16h a;
#pragma unroll
        for (int j = 0; j < 8; ++j) { a[j] = p1[j]; a[j + 8] = p2[j]; }
        const v16h b = *(const v16h*)(Wp + (((size_t)(kt * 2 + nt) * 32 + lane) << 4));
        c = __builtin_amdgcn_wmma_f32_16x16x32_f16(false, a, false, b, (short)0, c,
                                                   false, false);
    }
    const int oc = nt * 16 + (lane & 15);
    const int mb = (lane & 16) ? 8 : 0;
    const float bv = cb[oc];
#pragma unroll
    for (int rr = 0; rr < 8; ++rr) {
        int rowo = M0 + mb + rr;
        int bo = rowo / 121, to = rowo % 121;
        out[bo * 3872 + oc * 121 + to] = c[rr] + bv;   // matches reshape(B, 32*121)
    }
}

// ---- utility kernels ------------------------------------------------------
__global__ void k_zero(float* p, int n)
{
    int i = blockIdx.x * blockDim.x + threadIdx.x;
    if (i < n) p[i] = 0.f;
}

// dst[r, 0..ds) = { src[r, c] for c < F ; 0 for F <= c < ds }
__global__ void k_copy_pad(float* __restrict__ dst, const float* __restrict__ src,
                           int total, int fs, int ds_, int F)
{
    int idx = blockIdx.x * blockDim.x + threadIdx.x;
    if (idx >= total) return;
    int r = idx / ds_, c = idx - r * ds_;
    dst[idx] = (c < F) ? src[(size_t)r * fs + c] : 0.f;
}

// agg[dst] += x[src]: one wave32 per edge, feature = lane.
// x row read is one coalesced 128B transaction; the 32 atomics per edge hit
// consecutive addresses (coalesced burst into the L2 atomic units).
__global__ void k_scatter(const float* __restrict__ x, int sx,
                          const int* __restrict__ src, const int* __restrict__ dst,
                          float* agg, int sa, int E, int F)
{
    int wid  = (blockIdx.x * blockDim.x + threadIdx.x) >> 5;   // edge = global wave
    int lane = threadIdx.x & 31;
    if (wid >= E) return;
    int s = src[wid], d = dst[wid];
    const float* xr = x + (size_t)s * sx;
    float* ar = agg + (size_t)d * sa;
    for (int f = lane; f < F; f += 32)
        atomicAdd(&ar[f], xr[f]);
}

// per-feature sum / sumsq over N rows (F = 32 = wave width)
__global__ void __launch_bounds__(256)
k_bn_stats(const float* __restrict__ a, float* sum, float* sumsq, int N)
{
    __shared__ float ls[8][32];
    __shared__ float lq[8][32];
    const int f = threadIdx.x & 31;
    const int g = threadIdx.x >> 5;
    const int base = blockIdx.x * 2048;
    const int end  = (base + 2048 < N) ? base + 2048 : N;
    float s = 0.f, q = 0.f;
    for (int r = base + g; r < end; r += 8) {
        float v = a[(size_t)r * 32 + f];
        s += v; q += v * v;
    }
    ls[g][f] = s; lq[g][f] = q;
    __syncthreads();
    if (g == 0) {
        for (int t = 1; t < 8; ++t) { s += ls[t][f]; q += lq[t][f]; }
        atomicAdd(&sum[f], s);
        atomicAdd(&sumsq[f], q);
    }
}

__global__ void k_bn_final(const float* __restrict__ sum, const float* __restrict__ sq,
                           const float* __restrict__ g, const float* __restrict__ beta,
                           float* ss, float invN)
{
    int f = threadIdx.x;
    if (f < 32) {
        float mu  = sum[f] * invN;
        float var = sq[f] * invN - mu * mu;
        float sc  = g[f] * rsqrtf(var + 1e-5f);
        ss[f]      = sc;
        ss[32 + f] = beta[f] - mu * sc;
    }
}

__global__ void k_bn_apply(const float* __restrict__ a, const float* __restrict__ ss,
                           float* __restrict__ out, int total)
{
    int i = blockIdx.x * blockDim.x + threadIdx.x;
    if (i >= total) return;
    int f = i & 31;
    out[i] = a[i] * ss[f] + ss[32 + f];
}

__global__ void k_pool(const float* __restrict__ h, const int* __restrict__ batch,
                       float* psum, float* cnt, int total)
{
    int idx = blockIdx.x * blockDim.x + threadIdx.x;
    if (idx >= total) return;
    int n = idx >> 5, f = idx & 31;
    int b = batch[n];
    atomicAdd(&psum[b * 32 + f], h[idx]);
    if (f == 0) atomicAdd(&cnt[b], 1.0f);
}

__global__ void k_pool_div(float* psum, const float* __restrict__ cnt, int B)
{
    int idx = blockIdx.x * blockDim.x + threadIdx.x;
    if (idx >= B * 32) return;
    float c = cnt[idx >> 5];
    psum[idx] /= fmaxf(c, 1.0f);
}

__global__ void k_final(const float* __restrict__ z2, const float* __restrict__ w3,
                        const float* __restrict__ b3, const float* __restrict__ y,
                        float* __restrict__ out, int B)
{
    int b = blockIdx.x * blockDim.x + threadIdx.x;
    if (b >= B) return;
    float acc = b3[0];
    for (int j = 0; j < 256; ++j) acc += z2[b * 256 + j] * w3[j];
    out[b]     = acc;
    out[B + b] = y[b];
}

// ---------------------------------------------------------------------------
extern "C" void kernel_launch(void* const* d_in, const int* in_sizes, int n_in,
                              void* d_out, int out_size, void* d_ws, size_t ws_size,
                              hipStream_t stream)
{
    const int*   xd      = (const int*)d_in[0];
    const float* xt      = (const float*)d_in[1];
    const int*   ei      = (const int*)d_in[2];
    const int*   xbatch  = (const int*)d_in[3];
    const float* y       = (const float*)d_in[4];
    const float* emb     = (const float*)d_in[5];
    const float* conv_w  = (const float*)d_in[6];
    const float* conv_b  = (const float*)d_in[7];
    const float* fc1xd_w = (const float*)d_in[8];
    const float* fc1xd_b = (const float*)d_in[9];
    const float* fc1xt_w = (const float*)d_in[40];
    const float* fc1xt_b = (const float*)d_in[41];
    const float* cls_w1  = (const float*)d_in[42];
    const float* cls_b1  = (const float*)d_in[43];
    const float* cls_w2  = (const float*)d_in[44];
    const float* cls_b2  = (const float*)d_in[45];
    const float* cls_w3  = (const float*)d_in[46];
    const float* cls_b3  = (const float*)d_in[47];
    float* out = (float*)d_out;

    const int B  = in_sizes[0] / 100;       // 512
    const int N  = in_sizes[1] / 41;        // 262144
    const int E  = in_sizes[2] / 2;         // 4194304
    const int FX = 41, DIM = 32;
    const int S0 = 64;                      // layer-0 padded K stride (Kp of 41)

    // ---- workspace carve-out ----
    size_t off = 0;
    auto alloc = [&](size_t bytes) -> void* {
        void* p = (char*)d_ws + off;
        off += (bytes + 255) & ~(size_t)255;
        return p;
    };
    _Float16* E16    = (_Float16*)alloc((size_t)B * 100 * 128 * 2);
    float*    convout= (float*)alloc((size_t)B * 3872 * 4);
    float*    xj     = (float*)alloc((size_t)B * 256 * 4);
    float*    z1     = (float*)alloc((size_t)B * 1024 * 4);
    float*    z2     = (float*)alloc((size_t)B * 256 * 4);
    float*    agg    = (float*)alloc((size_t)N * S0 * 4);   // padded, layer0 uses 64
    float*    t1     = (float*)alloc((size_t)N * DIM * 4);
    float*    ab     = (float*)alloc((size_t)N * DIM * 4);
    float*    hA     = (float*)alloc((size_t)N * DIM * 4);
    float*    hB     = (float*)alloc((size_t)N * DIM * 4);
    float*    bnbuf  = (float*)alloc(64 * 4);
    float*    ssbuf  = (float*)alloc(64 * 4);
    float*    psum   = (float*)alloc((size_t)B * 32 * 4);
    float*    cnt    = (float*)alloc((size_t)B * 4);
    // pre-packed f16 weights (fragment order)
    _Float16* pk_conv = (_Float16*)alloc((size_t)800 * 32 * 2);
    _Float16* pk_fcxd = (_Float16*)alloc((size_t)3872 * 128 * 2);
    _Float16* pk_w1[5], *pk_w2[5];
    for (int l = 0; l < 5; ++l) {
        pk_w1[l] = (_Float16*)alloc((size_t)64 * 32 * 2);   // Kp=64 (l0) / 32
        pk_w2[l] = (_Float16*)alloc((size_t)32 * 32 * 2);
    }
    _Float16* pk_fcxt = (_Float16*)alloc((size_t)32 * 128 * 2);
    _Float16* pk_c1   = (_Float16*)alloc((size_t)256 * 1024 * 2);
    _Float16* pk_c2   = (_Float16*)alloc((size_t)1024 * 256 * 2);
    (void)ws_size; (void)n_in; (void)out_size;

    // ---- weight pre-packing ----
    auto prepack = [&](const float* W, _Float16* dst, int K, int Nn, int conv) {
        int total = (((K + 31) >> 5) << 5) * Nn;
        k_prepack<<<CDIV(total, 256), 256, 0, stream>>>(W, dst, K, Nn, total, conv);
    };
    prepack(conv_w, pk_conv, 800, 32, 1);
    prepack(fc1xd_w, pk_fcxd, 3872, 128, 0);
    for (int l = 0; l < 5; ++l) {
        prepack((const float*)d_in[10 + 6 * l + 0], pk_w1[l], l == 0 ? FX : DIM, 32, 0);
        prepack((const float*)d_in[10 + 6 * l + 2], pk_w2[l], DIM, 32, 0);
    }
    prepack(fc1xt_w, pk_fcxt, 32, 128, 0);
    prepack(cls_w1, pk_c1, 256, 1024, 0);
    prepack(cls_w2, pk_c2, 1024, 256, 0);

    // ================= drug branch =================
    {
        int total = B * 100 * 128;
        k_embed<<<CDIV(total, 256), 256, 0, stream>>>(xd, emb, E16, total);
        k_conv_gemm<<<dim3(CDIV(B * 121, 64), 1), 256, 0, stream>>>(E16, pk_conv,
                                                                    conv_b, convout);
        // hd = convout @ fc1_xd_w + b  ->  xj[:, 0:128]  (no activation)
        k_gemm<<<dim3(B / 64, 128 / 32), 256, 0, stream>>>(
            convout, 3872, pk_fcxd, fc1xd_b, xj, 256, 128, 3872, 0);
    }

    // ================= protein GIN branch =================
    const float* hin = xt;
    int Fin = FX, sin = FX;                       // input feature count / row stride
    for (int l = 0; l < 5; ++l) {
        const float* b1   = (const float*)d_in[10 + 6 * l + 1];
        const float* b2   = (const float*)d_in[10 + 6 * l + 3];
        const float* gam  = (const float*)d_in[10 + 6 * l + 4];
        const float* beta = (const float*)d_in[10 + 6 * l + 5];
        const int sa = (l == 0) ? S0 : DIM;       // padded agg stride

        int total = N * sa;
        k_copy_pad<<<CDIV(total, 256), 256, 0, stream>>>(agg, hin, total, sin, sa, Fin);
        k_scatter<<<CDIV(E * 32, 256), 256, 0, stream>>>(hin, sin, ei, ei + E,
                                                         agg, sa, E, Fin);
        // relu(agg @ w1 + b1)
        k_gemm<<<dim3(N / 64, 1), 256, 0, stream>>>(agg, sa, pk_w1[l], b1,
                                                    t1, DIM, DIM, Fin, 1);
        // relu(t1 @ w2 + b2)   (relu-before-BN per reference)
        k_gemm<<<dim3(N / 64, 1), 256, 0, stream>>>(t1, DIM, pk_w2[l], b2,
                                                    ab, DIM, DIM, DIM, 1);
        // batch-norm over node dim
        k_zero<<<1, 64, 0, stream>>>(bnbuf, 64);
        k_bn_stats<<<CDIV(N, 2048), 256, 0, stream>>>(ab, bnbuf, bnbuf + 32, N);
        k_bn_final<<<1, 32, 0, stream>>>(bnbuf, bnbuf + 32, gam, beta, ssbuf,
                                         1.0f / (float)N);
        float* hout = (l & 1) ? hB : hA;
        k_bn_apply<<<CDIV(N * DIM, 256), 256, 0, stream>>>(ab, ssbuf, hout, N * DIM);
        hin = hout; Fin = DIM; sin = DIM;
    }

    // global mean pool
    k_zero<<<CDIV(B * 32, 256), 256, 0, stream>>>(psum, B * 32);
    k_zero<<<CDIV(B, 256), 256, 0, stream>>>(cnt, B);
    k_pool<<<CDIV(N * 32, 256), 256, 0, stream>>>(hin, xbatch, psum, cnt, N * 32);
    k_pool_div<<<CDIV(B * 32, 256), 256, 0, stream>>>(psum, cnt, B);
    // ht = relu(pool @ fc1_xt_w + b) -> xj[:, 128:256]
    k_gemm<<<dim3(B / 64, 128 / 32), 256, 0, stream>>>(
        psum, 32, pk_fcxt, fc1xt_b, xj + 128, 256, 128, 32, 1);

    // ================= classifier =================
    k_gemm<<<dim3(B / 64, 1024 / 32), 256, 0, stream>>>(
        xj, 256, pk_c1, cls_b1, z1, 1024, 1024, 256, 1);
    k_gemm<<<dim3(B / 64, 256 / 32), 256, 0, stream>>>(
        z1, 1024, pk_c2, cls_b2, z2, 256, 256, 1024, 1);
    k_final<<<CDIV(B, 256), 256, 0, stream>>>(z2, cls_w3, cls_b3, y, out, B);
}
